// TemporalFeatureExtractor_65377992180132
// MI455X (gfx1250) — compile-verified, matrix operand order checked by reading, not run on hardware
//
#include <hip/hip_runtime.h>

// LSTM: B=256, T=128, F=1280, H=256.  Inputs (f32): x[B,T,F], kernel[F,4H],
// recurrent_kernel[H,4H], bias[4H].  Output: h_last[B,H] f32.
//
// Phase 1: xz = x @ kernel + bias  via v_wmma_f32_16x16x32_f16 (f16 in, f32 acc)
// Phase 2: sequential recurrence; h@R via WMMA (gate-aligned wave tiling) with
//          the per-step xz slab DMA'd into LDS by the Tensor Data Mover,
//          overlapped with the WMMA phase (tensor_load_to_lds + s_wait_tensorcnt).

typedef __attribute__((ext_vector_type(16))) _Float16 v16h;
typedef __attribute__((ext_vector_type(8)))  _Float16 v8h;
typedef __attribute__((ext_vector_type(8)))  float    v8f;
typedef __attribute__((ext_vector_type(4)))  unsigned int v4u;
typedef __attribute__((ext_vector_type(8)))  int      v8i;
typedef __attribute__((ext_vector_type(4)))  int      v4i;

#define B_  256
#define T_  128
#define F_  1280
#define H_  256
#define G4  1024   // 4*H

static __device__ __forceinline__ v16h make_frag(v8h lo, v8h hi) {
  return __builtin_shufflevector(lo, hi, 0,1,2,3,4,5,6,7,8,9,10,11,12,13,14,15);
}

static __device__ __forceinline__ float sigmoidf_(float v) {
  return 1.0f / (1.0f + __expf(-v));
}

// ---------------------------------------------------------------------------
// TDM: async 2D f32 tile load Global -> LDS.  Descriptor per CDNA5 ISA §8.3-8.4:
//   group0: count=1 | lds_addr | global_addr(57b) | type=2
//   group1: data_size=2(4B) | tensor_dim0/1 | tile_dim0/1 | dim0_stride(48b)
// Toolchain arity differs (5-arg ROCm vs 6-arg therock); the TDM header only
// ships with the 6-arg toolchain, so use it as the arity probe.
// ---------------------------------------------------------------------------
#if __has_include(<hip/amd_detail/amd_gfx1250_TDM.h>)
#define TDM_6ARG 1
#else
#define TDM_6ARG 0
#endif

static __device__ __forceinline__ void tdm_load_2d_f32(
    unsigned lds_off, const float* gptr,
    unsigned tile_d0, unsigned tile_d1, unsigned long long row_stride_elems) {
  unsigned long long ga = (unsigned long long)(size_t)gptr;
  v4u g0;
  g0.x = 1u;                                               // count=1 (valid user D#)
  g0.y = lds_off;                                          // lds_addr (bytes)
  g0.z = (unsigned)(ga & 0xffffffffu);                     // global_addr[31:0]
  g0.w = (unsigned)((ga >> 32) & 0x1ffffffu) | (2u << 30); // addr[56:32] | type=2
  v8i g1;
  unsigned long long s0 = row_stride_elems;                // data_size units
  g1[0] = (int)(2u << 16);                                 // data_size=2 -> 4 bytes
  g1[1] = (int)((tile_d0 & 0xffffu) << 16);                // tensor_dim0[15:0]
  g1[2] = (int)((tile_d0 >> 16) | ((tile_d1 & 0xffffu) << 16)); // dim0 hi | dim1 lo
  g1[3] = (int)((tile_d1 >> 16) | ((tile_d0 & 0xffffu) << 16)); // dim1 hi | tile_dim0
  g1[4] = (int)(tile_d1 & 0xffffu);                        // tile_dim1 (tile_dim2=0)
  g1[5] = (int)(s0 & 0xffffffffu);                         // tensor_dim0_stride lo
  g1[6] = (int)((s0 >> 32) & 0xffffu);                     // stride hi (dim1_stride=0)
  g1[7] = 0;
  v4i z4 = {0, 0, 0, 0};
#if TDM_6ARG
  v8i z8 = {0, 0, 0, 0, 0, 0, 0, 0};
  __builtin_amdgcn_tensor_load_to_lds(g0, g1, z4, z4, z8, 0);
#else
  __builtin_amdgcn_tensor_load_to_lds(g0, g1, z4, z4, 0);
#endif
}

// ---------------------------------------------------------------------------
// Kernel 1: xz[M=32768, N=1024] = x[M, K=1280] @ Wk[K, N] + bias[N]
// WG: 128 threads (4 waves), 64x64 tile; each wave does a 32x32 quadrant
// (2x2 WMMA 16x16x32 tiles). K staged in 32-wide f16 LDS tiles with b128
// global loads and packed f32->f16 conversion on store.
// ---------------------------------------------------------------------------
__global__ __launch_bounds__(128)
void lstm_xz_gemm(const float* __restrict__ x, const float* __restrict__ Wk,
                  const float* __restrict__ bias, float* __restrict__ xz) {
  __shared__ _Float16 As[64][32];   // [m][k]
  __shared__ _Float16 Bs[64][32];   // [n][k]  (transposed while staging)

  const int tid  = threadIdx.x;
  const int lane = tid & 31;
  const int wave = tid >> 5;        // 0..3
  const int wm   = wave >> 1;       // 0..1 (M half)
  const int wn   = wave & 1;        // 0..1 (N half)
  const int m0   = blockIdx.x * 64; // over 32768
  const int n0   = blockIdx.y * 64; // over 1024

  const int lr     = lane & 15;
  const int kc     = (lane >> 4) << 3;    // K sub-chunk base: 0 or 8
  const int rowoff = (lane >> 4) << 3;    // C-layout row offset: 0 or 8

  // Per-thread staging coordinates
  const int ar  = tid >> 1;               // A row 0..63
  const int ak0 = (tid & 1) * 16;         // A col 0 or 16
  const int bk  = tid >> 2;               // B k 0..31
  const int bn0 = (tid & 3) * 16;         // B n 0,16,32,48

  v8f acc[2][2] = {};

  for (int k0 = 0; k0 < F_; k0 += 32) {
    if (k0 + 32 < F_)
      __builtin_prefetch(&x[(size_t)(m0 + ar) * F_ + k0 + 32], 0, 1);

    __syncthreads();
    {   // Stage A: 16 contiguous f32 -> 16 f16 per thread
      const float4* xp = (const float4*)&x[(size_t)(m0 + ar) * F_ + k0 + ak0];
      float4 f0 = xp[0], f1 = xp[1], f2 = xp[2], f3 = xp[3];
      v8h h0 = {(_Float16)f0.x, (_Float16)f0.y, (_Float16)f0.z, (_Float16)f0.w,
                (_Float16)f1.x, (_Float16)f1.y, (_Float16)f1.z, (_Float16)f1.w};
      v8h h1 = {(_Float16)f2.x, (_Float16)f2.y, (_Float16)f2.z, (_Float16)f2.w,
                (_Float16)f3.x, (_Float16)f3.y, (_Float16)f3.z, (_Float16)f3.w};
      *(v8h*)&As[ar][ak0]     = h0;
      *(v8h*)&As[ar][ak0 + 8] = h1;
    }
    {   // Stage B transposed: 16 contiguous f32 along n, scattered f16 stores
      const float4* wp = (const float4*)&Wk[(size_t)(k0 + bk) * G4 + n0 + bn0];
      float4 f0 = wp[0], f1 = wp[1], f2 = wp[2], f3 = wp[3];
      float v[16] = {f0.x, f0.y, f0.z, f0.w, f1.x, f1.y, f1.z, f1.w,
                     f2.x, f2.y, f2.z, f2.w, f3.x, f3.y, f3.z, f3.w};
      #pragma unroll
      for (int i = 0; i < 16; ++i) Bs[bn0 + i][bk] = (_Float16)v[i];
    }
    __syncthreads();

    // Fragments per ISA 16-bit A layout (lane<16: K 0..7 & 16..23, else 8..15 & 24..31)
    v16h a0 = make_frag(*(const v8h*)&As[wm * 32 + lr][kc],
                        *(const v8h*)&As[wm * 32 + lr][kc + 16]);
    v16h a1 = make_frag(*(const v8h*)&As[wm * 32 + 16 + lr][kc],
                        *(const v8h*)&As[wm * 32 + 16 + lr][kc + 16]);
    v16h b0 = make_frag(*(const v8h*)&Bs[wn * 32 + lr][kc],
                        *(const v8h*)&Bs[wn * 32 + lr][kc + 16]);
    v16h b1 = make_frag(*(const v8h*)&Bs[wn * 32 + 16 + lr][kc],
                        *(const v8h*)&Bs[wn * 32 + 16 + lr][kc + 16]);

    acc[0][0] = __builtin_amdgcn_wmma_f32_16x16x32_f16(false, a0, false, b0, (short)0, acc[0][0], false, false);
    acc[0][1] = __builtin_amdgcn_wmma_f32_16x16x32_f16(false, a0, false, b1, (short)0, acc[0][1], false, false);
    acc[1][0] = __builtin_amdgcn_wmma_f32_16x16x32_f16(false, a1, false, b0, (short)0, acc[1][0], false, false);
    acc[1][1] = __builtin_amdgcn_wmma_f32_16x16x32_f16(false, a1, false, b1, (short)0, acc[1][1], false, false);
  }

  // Epilogue: C layout VGPR r -> row (r + 8*(lane>>4)), col (lane&15); fuse bias
  #pragma unroll
  for (int mt = 0; mt < 2; ++mt) {
    #pragma unroll
    for (int nt = 0; nt < 2; ++nt) {
      int n = n0 + wn * 32 + nt * 16 + lr;
      float bv = bias[n];
      int mbase = m0 + wm * 32 + mt * 16 + rowoff;
      #pragma unroll
      for (int r = 0; r < 8; ++r)
        xz[(size_t)(mbase + r) * G4 + n] = acc[mt][nt][r] + bv;
    }
  }
}

// ---------------------------------------------------------------------------
// Kernel 2a: Rt[n][k] (f16) = recurrent_kernel[k][n] (f32)  — 512 KB, L2-resident
// ---------------------------------------------------------------------------
__global__ __launch_bounds__(256)
void convert_R(const float* __restrict__ R, _Float16* __restrict__ Rt) {
  int idx = blockIdx.x * 256 + threadIdx.x;  // over 1024*256
  int n = idx >> 8;
  int k = idx & 255;
  Rt[idx] = (_Float16)R[(size_t)k * G4 + n];
}

// ---------------------------------------------------------------------------
// Kernel 2b: sequential recurrence.  16 WGs x 256 threads (8 waves); each WG
// owns 16 batch rows.  Wave w owns hidden columns [w*32, w*32+32) in ALL four
// gate blocks, so i/f/g/o for an element sit in the same lane -> cell state c
// lives in registers.  Per step, wave 0 issues a TDM descriptor that DMA's the
// 16x1024 f32 xz slab for timestep t into LDS while all waves run the h@R
// WMMAs; s_wait_tensorcnt + barrier fence it before the gate phase.
// ---------------------------------------------------------------------------
__global__ __launch_bounds__(256)
void lstm_recurrent(const float* __restrict__ xz, const _Float16* __restrict__ Rt,
                    float* __restrict__ out) {
  __shared__ _Float16 hbuf[16][H_];    // 8 KB  — h (f16), WMMA A operand
  __shared__ float    xzbuf[16][G4];   // 64 KB — TDM destination for xz slab

  const int tid  = threadIdx.x;
  const int lane = tid & 31;
  const int wave = tid >> 5;          // 0..7
  const int b0   = blockIdx.x * 16;   // batch tile base

  const int lr     = lane & 15;
  const int kc     = (lane >> 4) << 3;
  const int rowoff = (lane >> 4) << 3;

  const unsigned xz_lds_off = (unsigned)(size_t)&xzbuf[0][0];

  for (int i = tid; i < 16 * H_; i += 256) hbuf[0][i] = (_Float16)0.0f;
  __syncthreads();

  float c[2][8];
  #pragma unroll
  for (int j = 0; j < 2; ++j)
    #pragma unroll
    for (int r = 0; r < 8; ++r) c[j][r] = 0.0f;

  for (int t = 0; t < T_; ++t) {
    if (wave == 0) {
      // Async DMA: rows = batch b0..b0+15 of xz[:, t, :]; row stride T*4H elems
      tdm_load_2d_f32(xz_lds_off, xz + ((size_t)b0 * T_ + t) * G4,
                      G4, 16, (unsigned long long)T_ * G4);
    }

    v8f acc[4][2] = {};   // [gate][sub-tile]

    #pragma unroll
    for (int kt = 0; kt < 8; ++kt) {
      int k0 = kt * 32;
      v16h a = make_frag(*(const v8h*)&hbuf[lr][k0 + kc],
                         *(const v8h*)&hbuf[lr][k0 + kc + 16]);
      #pragma unroll
      for (int g = 0; g < 4; ++g) {
        #pragma unroll
        for (int j = 0; j < 2; ++j) {
          int n = g * 256 + wave * 32 + j * 16 + lr;
          const _Float16* bp = Rt + (size_t)n * H_ + k0;
          v16h bf = make_frag(*(const v8h*)(bp + kc), *(const v8h*)(bp + kc + 16));
          acc[g][j] = __builtin_amdgcn_wmma_f32_16x16x32_f16(
              false, a, false, bf, (short)0, acc[g][j], false, false);
        }
      }
    }

    if (wave == 0) __builtin_amdgcn_s_wait_tensorcnt(0);  // TDM slab landed
    __syncthreads();   // hbuf reads done + xzbuf visible to all waves

    #pragma unroll
    for (int j = 0; j < 2; ++j) {
      int nH = wave * 32 + j * 16 + lr;
      #pragma unroll
      for (int r = 0; r < 8; ++r) {
        int row = r + rowoff;                                  // 0..15
        float zi = acc[0][j][r] + xzbuf[row][nH];
        float zf = acc[1][j][r] + xzbuf[row][256 + nH];
        float zg = acc[2][j][r] + xzbuf[row][512 + nH];
        float zo = acc[3][j][r] + xzbuf[row][768 + nH];
        float ig = sigmoidf_(zi);
        float fg = sigmoidf_(zf);
        float gg = tanhf(zg);
        float og = sigmoidf_(zo);
        float cn = fg * c[j][r] + ig * gg;
        c[j][r] = cn;
        float hn = og * tanhf(cn);
        hbuf[row][nH] = (_Float16)hn;
        if (t == T_ - 1)
          out[(size_t)(b0 + row) * H_ + nH] = hn;
      }
    }
    __syncthreads();   // h/xz consumption done before next step's TDM + A loads
  }
}

// ---------------------------------------------------------------------------
extern "C" void kernel_launch(void* const* d_in, const int* in_sizes, int n_in,
                              void* d_out, int out_size, void* d_ws, size_t ws_size,
                              hipStream_t stream) {
  (void)in_sizes; (void)n_in; (void)out_size; (void)ws_size;
  const float* x    = (const float*)d_in[0];                 // [B,T,F]
  const float* Wk   = (const float*)d_in[1];                 // [F,4H]
  const float* R    = (const float*)d_in[2];                 // [H,4H]
  const float* bias = (const float*)d_in[3];                 // [4H]
  float* out = (float*)d_out;                                // [B,H]

  // Workspace layout: xz f32 [32768,1024] (128 MiB), then Rt f16 [1024,256]
  float*    xzw = (float*)d_ws;
  _Float16* Rt  = (_Float16*)((char*)d_ws + (size_t)B_ * T_ * G4 * sizeof(float));

  convert_R<<<(G4 * H_) / 256, 256, 0, stream>>>(R, Rt);

  dim3 g1((B_ * T_) / 64, G4 / 64);          // 512 x 16 workgroups
  lstm_xz_gemm<<<g1, 128, 0, stream>>>(x, Wk, bias, xzw);

  lstm_recurrent<<<B_ / 16, 256, 0, stream>>>(xzw, Rt, out);
}
